// LocalAttention_86517821216554
// MI455X (gfx1250) — matrix operand order repaired and checked by reading, not compiled, not run on hardware
//
#include <hip/hip_runtime.h>

// ---------------------------------------------------------------------------
// Problem constants (from reference): L=1024 seq, N=8 batch, E=512 embed,
// H=8 heads, Dh=64 head dim, local window +/-128.
// ---------------------------------------------------------------------------
constexpr int L  = 1024;
constexpr int Nb = 8;
constexpr int E  = 512;
constexpr int H  = 8;
constexpr int Dh = 64;
constexpr int W2 = 128;

typedef __bf16 bf16_t;
typedef bf16_t v16bf __attribute__((ext_vector_type(16)));
typedef bf16_t v8bf  __attribute__((ext_vector_type(8)));
typedef float  v8f   __attribute__((ext_vector_type(8)));

__device__ __forceinline__ v8f wmma_bf16(v16bf a, v16bf b, v8f c) {
  // D(f32 16x16) = A(bf16 16x32) * B(bf16 32x16) + C
  return __builtin_amdgcn_wmma_f32_16x16x32_bf16(
      /*neg_a=*/false, a, /*neg_b=*/false, b,
      /*c_mod=*/(short)0, c, /*reuse_a=*/false, /*reuse_b=*/false);
}

// ---------------------------------------------------------------------------
// Kernel 1: projection  P = X @ W^T + b   (X: [M=L*N, E] f32, W: [E,E] f32)
// Output stored bf16 in [N, H, L, Dh] layout for the scores GEMM.
// One wave per 16x16 output tile; K-loop unrolled x4 for wmma density.
// A-lane row = lane&15 of X; B-lane column (lane&15) is a row of W (since
// B(k,n) = W[n,k]), so K is contiguous for both fragments.
// ---------------------------------------------------------------------------
__global__ void proj_kernel(const float* __restrict__ X,
                            const float* __restrict__ W,
                            const float* __restrict__ bias,
                            bf16_t* __restrict__ P) {
  const int lane = threadIdx.x;
  const int g = lane >> 4, half = lane & 15;
  const int m0 = blockIdx.x * 16;
  const int e0 = blockIdx.y * 16;

  const float* xrow = X + (size_t)(m0 + half) * E;  // A: row of X
  const float* wrow = W + (size_t)(e0 + half) * E;  // B: column e = row of W

  v8f acc = {0.f, 0.f, 0.f, 0.f, 0.f, 0.f, 0.f, 0.f};
#pragma unroll 4
  for (int kk = 0; kk < E; kk += 32) {
    __builtin_prefetch(xrow + kk + 128, 0, 0);   // next slab (speculative)
    __builtin_prefetch(wrow + kk + 128, 0, 0);
    v16bf a, b;
#pragma unroll
    for (int i = 0; i < 8; ++i) {
      a[i]     = (bf16_t)xrow[kk + g * 8 + i];        // K = g*8 + i
      a[i + 8] = (bf16_t)xrow[kk + 16 + g * 8 + i];   // K = 16 + g*8 + i
    }
#pragma unroll
    for (int i = 0; i < 8; ++i) {
      b[i]     = (bf16_t)wrow[kk + g * 8 + i];        // K = g*8 + i
      b[i + 8] = (bf16_t)wrow[kk + 16 + g * 8 + i];   // K = 16 + g*8 + i
    }
    acc = wmma_bf16(a, b, acc);
  }

  const int e  = e0 + half;
  const float bv = bias[e];
  const int hh = e >> 6, d = e & (Dh - 1);
#pragma unroll
  for (int v = 0; v < 8; ++v) {
    const int m = m0 + v + g * 8;       // row of X = l*N + n
    const int l = m >> 3, nb = m & (Nb - 1);
    P[(((size_t)nb * H + hh) * L + l) * Dh + d] = (bf16_t)(acc[v] + bv);
  }
}

// ---------------------------------------------------------------------------
// Kernel 2: scores + softmax + head-mean + local mask + renorm.
// Block = 128 threads (4 waves), owns (n, 16 query rows). Per head: each wave
// computes 16 score tiles (256 columns) with 2 WMMAs/tile into LDS sc[];
// softmax (max, exp stored back into sc, sum) and head-mean accumulation into
// LDS wacc[] (no global RMW traffic). Final pass applies +/-W2 mask and
// renormalizes, writing w_loc to global exactly once.
// LDS: 64KB sc + 64KB wacc + reduce scratch (~129KB of the 320KB/WGP).
// ---------------------------------------------------------------------------
__global__ void scores_kernel(const bf16_t* __restrict__ Qp,
                              const bf16_t* __restrict__ Kp,
                              float* __restrict__ Wout) {
  __shared__ float sc[16][L];      // raw scores -> exp values
  __shared__ float wacc[16][L];    // head-mean accumulator
  __shared__ float red[16][8];
  __shared__ float rowstat[16];

  const int tid  = threadIdx.x;
  const int lane = tid & 31, wave = tid >> 5;
  const int g = lane >> 4, half = lane & 15;
  const int l0 = blockIdx.x * 16;
  const int n  = blockIdx.y;

  const int r   = tid >> 3;             // softmax row 0..15
  const int seg = tid & 7;              // interleaved column lane within row
  const float scale = 0.125f;           // 1/sqrt(Dh)
  const float invH  = 1.0f / (float)H;

  // zero the accumulator (interleaved: consecutive seg -> consecutive banks)
  for (int k = 0; k < L / 8; ++k) wacc[r][seg + 8 * k] = 0.f;

  for (int h = 0; h < H; ++h) {
    const size_t base = ((size_t)n * H + h) * L;

    // Q fragments for this wave's lane (reused across all 16 m-tiles)
    const bf16_t* qrow = Qp + (base + l0 + half) * Dh;
    v16bf aq[2];
#pragma unroll
    for (int kb = 0; kb < 2; ++kb) {
      const bf16_t* p = qrow + kb * 32;
      v8bf lo = *(const v8bf*)(p + g * 8);
      v8bf hi = *(const v8bf*)(p + 16 + g * 8);
#pragma unroll
      for (int i = 0; i < 8; ++i) { aq[kb][i] = lo[i]; aq[kb][i + 8] = hi[i]; }
    }

#pragma unroll 2
    for (int j = 0; j < 16; ++j) {
      const int m0t = wave * 256 + j * 16;
      const bf16_t* kcol = Kp + (base + m0t + half) * Dh;  // B column = key row
      v8f acc = {0.f, 0.f, 0.f, 0.f, 0.f, 0.f, 0.f, 0.f};
#pragma unroll
      for (int kb = 0; kb < 2; ++kb) {
        v16bf b;
        v8bf lo = *(const v8bf*)(kcol + kb * 32 + g * 8);
        v8bf hi = *(const v8bf*)(kcol + kb * 32 + 16 + g * 8);
#pragma unroll
        for (int i = 0; i < 8; ++i) { b[i] = lo[i]; b[i + 8] = hi[i]; }
        acc = wmma_bf16(aq[kb], b, acc);
      }
#pragma unroll
      for (int v = 0; v < 8; ++v)
        sc[v + g * 8][m0t + half] = acc[v] * scale;
    }
    __syncthreads();

    // row max (interleaved columns to avoid bank serialization)
    float mx = -3.0e38f;
    for (int k = 0; k < L / 8; ++k) mx = fmaxf(mx, sc[r][seg + 8 * k]);
    red[r][seg] = mx;
    __syncthreads();
    if (seg == 0) {
      float m = red[r][0];
#pragma unroll
      for (int q = 1; q < 8; ++q) m = fmaxf(m, red[r][q]);
      rowstat[r] = m;
    }
    __syncthreads();
    const float rm = rowstat[r];

    // exp (stored back into sc) + partial sum
    float s = 0.f;
    for (int k = 0; k < L / 8; ++k) {
      const int c = seg + 8 * k;
      const float e = __expf(sc[r][c] - rm);
      sc[r][c] = e;
      s += e;
    }
    red[r][seg] = s;
    __syncthreads();
    if (seg == 0) {
      float t = 0.f;
#pragma unroll
      for (int q = 0; q < 8; ++q) t += red[r][q];
      rowstat[r] = 1.0f / t;
    }
    __syncthreads();

    // accumulate head-mean attention into LDS accumulator
    const float inv = rowstat[r] * invH;
    for (int k = 0; k < L / 8; ++k) {
      const int c = seg + 8 * k;
      wacc[r][c] += sc[r][c] * inv;
    }
    __syncthreads();  // sc reused next head
  }

  // local-window mask + renormalize, single global write
  const int l = l0 + r;
  const int clo = (l - W2 < 0) ? 0 : l - W2;
  const int chi = (l + W2 > L - 1) ? L - 1 : l + W2;
  float s = 0.f;
  for (int k = 0; k < L / 8; ++k) {
    const int c = seg + 8 * k;
    if (c >= clo && c <= chi) s += wacc[r][c];
  }
  red[r][seg] = s;
  __syncthreads();
  if (seg == 0) {
    float t = 0.f;
#pragma unroll
    for (int q = 0; q < 8; ++q) t += red[r][q];
    rowstat[r] = 1.0f / (t + 1e-6f);
  }
  __syncthreads();
  const float inv = rowstat[r];
  float* wrow_g = Wout + ((size_t)n * L + l) * L;
  for (int k = 0; k < L / 8; ++k) {
    const int c = seg + 8 * k;
    wrow_g[c] = (c >= clo && c <= chi) ? wacc[r][c] * inv : 0.f;
  }
}

// ---------------------------------------------------------------------------
// Kernel 3: out[l,n,e] = sum_m w_loc[n,l,m] * value[m,n,e], band-limited to
// the local window (w_loc is exactly zero outside, so including the covering
// 32-aligned m-tiles is exact). One wave per 16x16 output tile; V tile staged
// through LDS (coalesced loads, then column gather for the B fragment).
// ---------------------------------------------------------------------------
__global__ void outgemm_kernel(const float* __restrict__ Wloc,
                               const float* __restrict__ V,
                               float* __restrict__ Out) {
  __shared__ bf16_t Vs[32][16];
  const int lane = threadIdx.x;
  const int g = lane >> 4, half = lane & 15;
  const int e0 = blockIdx.x * 16;
  const int l0 = blockIdx.y * 16;
  const int n  = blockIdx.z;

  int mstart = l0 - W2; if (mstart < 0) mstart = 0;
  mstart &= ~31;
  int mend = l0 + 16 + W2; if (mend > L) mend = L;

  const float* wrow = Wloc + ((size_t)n * L + l0 + half) * L;
  v8f acc = {0.f, 0.f, 0.f, 0.f, 0.f, 0.f, 0.f, 0.f};

  const int cc  = lane & 15;       // staging column (e offset)
  const int rr0 = lane >> 4;       // staging start row (m offset)

  for (int m0 = mstart; m0 < mend; m0 += 32) {
    // stage 32(m) x 16(e) tile of V as bf16: 32 lanes cover 2 rows per step
#pragma unroll
    for (int i = 0; i < 16; ++i) {
      const int rr = rr0 + 2 * i;
      const float* vp = V + ((size_t)(m0 + rr) * Nb + n) * E + e0 + cc;
      __builtin_prefetch(vp + 32 * Nb * E, 0, 0);  // next m-tile (speculative)
      Vs[rr][cc] = (bf16_t)(*vp);
    }
    __syncthreads();

    v16bf a, b;
#pragma unroll
    for (int i = 0; i < 8; ++i) {
      a[i]     = (bf16_t)wrow[m0 + g * 8 + i];
      a[i + 8] = (bf16_t)wrow[m0 + 16 + g * 8 + i];
    }
#pragma unroll
    for (int i = 0; i < 8; ++i) {
      b[i]     = Vs[g * 8 + i][half];       // B(k,n): k = g*8+i, col = half
      b[i + 8] = Vs[16 + g * 8 + i][half];  // k = 16+g*8+i
    }
    acc = wmma_bf16(a, b, acc);
    __syncthreads();
  }

#pragma unroll
  for (int v = 0; v < 8; ++v) {
    const int l = l0 + v + g * 8;
    Out[((size_t)l * Nb + n) * E + e0 + half] = acc[v];
  }
}

// ---------------------------------------------------------------------------
// Launch
// ---------------------------------------------------------------------------
extern "C" void kernel_launch(void* const* d_in, const int* in_sizes, int n_in,
                              void* d_out, int out_size, void* d_ws, size_t ws_size,
                              hipStream_t stream) {
  (void)in_sizes; (void)n_in; (void)out_size; (void)ws_size;
  const float* query = (const float*)d_in[0];  // [L,N,E]
  const float* key   = (const float*)d_in[1];  // [L,N,E]
  const float* value = (const float*)d_in[2];  // [L,N,E]
  const float* Wq    = (const float*)d_in[3];  // [E,E]
  const float* Wk    = (const float*)d_in[4];  // [E,E]
  // d_in[5] = Wv, d_in[8] = bv: unused (v-projection output is discarded)
  const float* bq    = (const float*)d_in[6];  // [E]
  const float* bk    = (const float*)d_in[7];  // [E]

  float* out  = (float*)d_out;                       // [L,N,E]
  float* wloc = out + (size_t)L * Nb * E;            // [N,L,L]

  bf16_t* Qp = (bf16_t*)d_ws;                        // [N,H,L,Dh] bf16
  bf16_t* Kp = Qp + (size_t)Nb * H * L * Dh;         // [N,H,L,Dh] bf16

  const dim3 gp(L * Nb / 16, E / 16);                // 512 x 32 tiles
  proj_kernel<<<gp, 32, 0, stream>>>(query, Wq, bq, Qp);
  proj_kernel<<<gp, 32, 0, stream>>>(key,   Wk, bk, Kp);

  scores_kernel<<<dim3(L / 16, Nb), 128, 0, stream>>>(Qp, Kp, wloc);

  outgemm_kernel<<<dim3(E / 16, L / 16, Nb), 32, 0, stream>>>(wloc, value, out);
}
